// Drift_75076028334152
// MI455X (gfx1250) — matrix operand order repaired and checked by reading, not compiled, not run on hardware
//
#include <hip/hip_runtime.h>

typedef __attribute__((ext_vector_type(16))) __bf16       v16bf;
typedef __attribute__((ext_vector_type(8)))  float        v8f;
typedef __attribute__((ext_vector_type(4)))  unsigned int v4u;

__device__ __forceinline__ unsigned short f2bf(float f) {
  unsigned int u = __float_as_uint(f);
  u += 0x7FFFu + ((u >> 16) & 1u);   // round-to-nearest-even
  return (unsigned short)(u >> 16);
}

union OpU { v4u u[2]; v16bf v; };

// A-operand (16x32 bf16): lane half holds contiguous 16 K values: [kofs,+16)
__device__ __forceinline__ v16bf lds_a16(const unsigned short* rowptr, int kofs) {
  OpU o;
  o.u[0] = *(const v4u*)(rowptr + kofs);
  o.u[1] = *(const v4u*)(rowptr + kofs + 8);
  return o.v;
}
// B-operand (32x16 bf16): lane half holds contiguous 16 K values: [kofs,+16)
__device__ __forceinline__ v16bf lds_b16(const unsigned short* rowptr, int kofs) {
  OpU o;
  o.u[0] = *(const v4u*)(rowptr + kofs);
  o.u[1] = *(const v4u*)(rowptr + kofs + 8);
  return o.v;
}

// Pass 0: f32 -> bf16 conversion + row squared norms.
__global__ void __launch_bounds__(64) drift_prep(const float* __restrict__ G,
                                                 const float* __restrict__ P,
                                                 unsigned short* __restrict__ Gb,
                                                 unsigned short* __restrict__ Pb,
                                                 float* __restrict__ tnorm) {
  const int row = blockIdx.x;               // 0..16383
  const int t   = threadIdx.x;              // 0..63, 4 floats each
  const float*  src = (row < 8192) ? (G + (size_t)row * 256) : (P + (size_t)(row - 8192) * 256);
  unsigned short* dst = (row < 8192) ? (Gb + (size_t)row * 256) : (Pb + (size_t)(row - 8192) * 256);
  float4 f = *(const float4*)(src + t * 4);
  unsigned int p0 = (unsigned int)f2bf(f.x) | ((unsigned int)f2bf(f.y) << 16);
  unsigned int p1 = (unsigned int)f2bf(f.z) | ((unsigned int)f2bf(f.w) << 16);
  *(unsigned int*)(dst + t * 4)     = p0;
  *(unsigned int*)(dst + t * 4 + 2) = p1;
  float ss = f.x * f.x + f.y * f.y + f.z * f.z + f.w * f.w;
  #pragma unroll
  for (int m = 16; m >= 1; m >>= 1) ss += __shfl_xor(ss, m, 32);
  __shared__ float red[2];
  if ((t & 31) == 0) red[t >> 5] = ss;
  __syncthreads();
  if (t == 0) tnorm[row] = red[0] + red[1];
}

// Fused main pass: per block, 64 output rows x 256 cols.
// out[i,:] = s_gen[i] * (k_pos[i,:] @ P)  -  s_pos[i] * (k_gen[i,:] @ G)
__global__ void __launch_bounds__(256, 1) drift_main(
    const unsigned short* __restrict__ Gb, const unsigned short* __restrict__ Pb,
    const float* __restrict__ tnorm, float* __restrict__ out) {
  __shared__ unsigned short Gs[64 * 256];        // block's G rows, bf16 (32 KB)
  __shared__ unsigned short Trow[2 * 32 * 256];  // j-tile row-major, double-buffered (32 KB)
  __shared__ unsigned short Ttr[2 * 256 * 32];   // j-tile transposed, double-buffered (32 KB)
  __shared__ unsigned short Karr[64 * 32];       // kernel values, bf16 (4 KB)
  __shared__ float sgen_sh[64];
  __shared__ float spos_sh[64];

  const int tid  = threadIdx.x;
  const int w    = tid >> 5;
  const int lane = tid & 31;
  const int l16  = lane & 15;
  const int hf   = lane >> 4;
  const int m16  = w >> 1;        // 16-row band 0..3
  const int n16  = w & 1;         // stage-1 j sub-tile
  const int dh   = w & 1;         // stage-2 D half (cols dh*128 .. +127)
  const int i0   = blockIdx.x * 64;

  // stage resident G tile into LDS
  {
    const v4u* s = (const v4u*)(Gb + (size_t)i0 * 256);
    v4u* d = (v4u*)Gs;
    #pragma unroll
    for (int k = 0; k < 8; ++k) d[tid + k * 256] = s[tid + k * 256];
  }
  if (tid < 64) { sgen_sh[tid] = 0.0f; spos_sh[tid] = 0.0f; }

  float gn[8];
  {
    const float* gp = tnorm + i0 + m16 * 16 + hf * 8;
    #pragma unroll
    for (int r = 0; r < 8; ++r) gn[r] = gp[r];
  }

  v8f acc1[8] = {};   // sum over j>=B of k_pos @ P  (D cols dh*128 + t*16 + l16)
  v8f acc2[8] = {};   // sum over j< B of k_gen @ G
  float sga[8] = {};  // per-lane partial row sums (gen)
  float spa[8] = {};  // per-lane partial row sums (pos)

  const unsigned short* arow = Gs   + (m16 * 16 + l16) * 256;
  const unsigned short* krow = Karr + (m16 * 16 + l16) * 32;

  const int jp = tid & 15;            // staging: j row pair
  const int cb = (tid >> 4) * 16;     // staging: 16-column slab

  for (int j0 = 0; j0 < 16384; j0 += 32) {
    const bool isg = (j0 < 8192);
    const unsigned short* src = isg ? (Gb + (size_t)j0 * 256)
                                    : (Pb + (size_t)(j0 - 8192) * 256);
    const int bsel = (j0 >> 5) & 1;
    unsigned short* TrowB = Trow + bsel * (32 * 256);
    unsigned short* TtrB  = Ttr  + bsel * (256 * 32);

    // --- stage j-tile into buffer bsel (other waves may still read bsel^1) ---
    {
      v4u r0a = *(const v4u*)(src + (2 * jp + 0) * 256 + cb);
      v4u r0b = *(const v4u*)(src + (2 * jp + 0) * 256 + cb + 8);
      v4u r1a = *(const v4u*)(src + (2 * jp + 1) * 256 + cb);
      v4u r1b = *(const v4u*)(src + (2 * jp + 1) * 256 + cb + 8);
      *(v4u*)(TrowB + (2 * jp + 0) * 256 + cb)     = r0a;
      *(v4u*)(TrowB + (2 * jp + 0) * 256 + cb + 8) = r0b;
      *(v4u*)(TrowB + (2 * jp + 1) * 256 + cb)     = r1a;
      *(v4u*)(TrowB + (2 * jp + 1) * 256 + cb + 8) = r1b;
      unsigned short ea[16], eb[16];
      *(v4u*)(ea) = r0a; *(v4u*)(ea + 8) = r0b;
      *(v4u*)(eb) = r1a; *(v4u*)(eb + 8) = r1b;
      #pragma unroll
      for (int c = 0; c < 16; ++c) {
        unsigned int pk = (unsigned int)ea[c] | ((unsigned int)eb[c] << 16);
        *(unsigned int*)(TtrB + (cb + c) * 32 + 2 * jp) = pk;
      }
    }
    // prefetch next j-tile into cache hierarchy (global_prefetch_b8)
    if (j0 + 32 < 16384) {
      const unsigned short* nsrc = (j0 + 32 < 8192)
          ? (Gb + (size_t)(j0 + 32) * 256)
          : (Pb + (size_t)(j0 + 32 - 8192) * 256);
      __builtin_prefetch(nsrc + (2 * jp + 0) * 256 + cb, 0, 1);
      __builtin_prefetch(nsrc + (2 * jp + 1) * 256 + cb, 0, 1);
    }
    __syncthreads();  // staging of buffer bsel complete

    // --- stage 1: S = G_tile(16x256) x T_tile^T(256x16), pipelined K chunks ---
    const unsigned short* brow = TrowB + (n16 * 16 + l16) * 256;
    v8f S = {};
    v16bf av = lds_a16(arow, hf * 16);
    v16bf bv = lds_b16(brow, hf * 16);
    #pragma unroll
    for (int kc = 0; kc < 8; ++kc) {
      v16bf an = av, bn = bv;
      if (kc < 7) {
        an = lds_a16(arow, (kc + 1) * 32 + hf * 16);
        bn = lds_b16(brow, (kc + 1) * 32 + hf * 16);
      }
      S = __builtin_amdgcn_wmma_f32_16x16x32_bf16(false, av, false, bv,
                                                  (short)0, S, false, false);
      av = an; bv = bn;
    }

    // --- distance -> kernel value transform ---
    const int   jcol = j0 + n16 * 16 + l16;
    const float tn   = tnorm[jcol];
    float kv[8];
    #pragma unroll
    for (int r = 0; r < 8; ++r) {
      const int irow = i0 + m16 * 16 + hf * 8 + r;
      float sq = gn[r] + tn - 2.0f * S[r];
      sq = sq > 0.0f ? sq : 0.0f;
      float dd = __builtin_sqrtf(sq);
      float kk = __expf(-0.05f * dd);            // exp(-dist/20)
      if (isg && (jcol == irow)) kk = 0.0f;      // diag: exp(-1e8/20) == 0
      kv[r] = kk;
      Karr[(m16 * 16 + hf * 8 + r) * 32 + n16 * 16 + l16] = f2bf(kk);
    }
    if (isg) {
      #pragma unroll
      for (int r = 0; r < 8; ++r) sga[r] += kv[r];
    } else {
      #pragma unroll
      for (int r = 0; r < 8; ++r) spa[r] += kv[r];
    }

    // preload first stage-2 B operand (Ttr already valid) before the barrier
    v16bf bm = lds_b16(TtrB + (dh * 128 + 0 * 16 + l16) * 32, hf * 16);
    __syncthreads();  // Karr writes complete

    // --- stage 2: acc += K(16x32) x M2(32x16 cols of G or P), pipelined ---
    v16bf ak = lds_a16(krow, hf * 16);
    if (isg) {
      #pragma unroll
      for (int t = 0; t < 8; ++t) {
        v16bf bn = bm;
        if (t < 7) bn = lds_b16(TtrB + (dh * 128 + (t + 1) * 16 + l16) * 32, hf * 16);
        acc2[t] = __builtin_amdgcn_wmma_f32_16x16x32_bf16(false, ak, false, bm,
                                                          (short)0, acc2[t], false, false);
        bm = bn;
      }
    } else {
      #pragma unroll
      for (int t = 0; t < 8; ++t) {
        v16bf bn = bm;
        if (t < 7) bn = lds_b16(TtrB + (dh * 128 + (t + 1) * 16 + l16) * 32, hf * 16);
        acc1[t] = __builtin_amdgcn_wmma_f32_16x16x32_bf16(false, ak, false, bm,
                                                          (short)0, acc1[t], false, false);
        bm = bn;
      }
    }
  }

  // --- combine row sums: 16-lane butterfly, then cross-wave via LDS atomics ---
  #pragma unroll
  for (int r = 0; r < 8; ++r) {
    float sg = sga[r], sp = spa[r];
    #pragma unroll
    for (int m = 1; m <= 8; m <<= 1) {
      sg += __shfl_xor(sg, m, 32);
      sp += __shfl_xor(sp, m, 32);
    }
    if (l16 == 0) {
      atomicAdd(&sgen_sh[m16 * 16 + hf * 8 + r], sg);
      atomicAdd(&spos_sh[m16 * 16 + hf * 8 + r], sp);
    }
  }
  __syncthreads();

  // --- finalize: out = s_gen*acc1 - s_pos*acc2 ---
  #pragma unroll
  for (int t = 0; t < 8; ++t) {
    #pragma unroll
    for (int r = 0; r < 8; ++r) {
      const int rl = m16 * 16 + hf * 8 + r;
      float v = sgen_sh[rl] * acc1[t][r] - spos_sh[rl] * acc2[t][r];
      out[(size_t)(i0 + rl) * 256 + dh * 128 + t * 16 + l16] = v;
    }
  }
}

extern "C" void kernel_launch(void* const* d_in, const int* in_sizes, int n_in,
                              void* d_out, int out_size, void* d_ws, size_t ws_size,
                              hipStream_t stream) {
  (void)in_sizes; (void)n_in; (void)out_size; (void)ws_size;
  const float* G = (const float*)d_in[0];   // (8192, 256) f32
  const float* P = (const float*)d_in[1];   // (8192, 256) f32
  unsigned short* Gb = (unsigned short*)d_ws;                 // 4 MiB bf16
  unsigned short* Pb = Gb + (size_t)8192 * 256;               // 4 MiB bf16
  float* tnorm = (float*)(Pb + (size_t)8192 * 256);           // 64 KiB f32 (||g||^2 then ||p||^2)
  float* out = (float*)d_out;

  drift_prep<<<16384, 64, 0, stream>>>(G, P, Gb, Pb, tnorm);
  drift_main<<<128, 256, 0, stream>>>(Gb, Pb, tnorm, out);
}
